// ArcticMoE_59562606461524
// MI455X (gfx1250) — matrix operand order repaired and checked by reading, not compiled, not run on hardware
//
#include <hip/hip_runtime.h>
#include <hip/hip_bf16.h>

// ---------------------------------------------------------------------------
// MoE (Arctic-style): T=8192 tokens, H=2048 hidden, I=4096 inter, E=8, top-2.
// MI455X (gfx1250) strategy:
//   - route tokens, compute ONLY selected (expert,token) pairs
//   - bf16 WMMA (v_wmma_f32_16x16x32_bf16); fp32 weights converted on the fly
//     (each expert's fp32 slice is L2-resident: 67MB vs 192MB L2)
//   - double-buffered LDS; A tiles staged with GLOBAL_LOAD_ASYNC_TO_LDS_B128
//     (ASYNCcnt path, overlapped with WMMA), B tiles via VALU cvt + ds_store
// ---------------------------------------------------------------------------

typedef __attribute__((ext_vector_type(8)))  float  v8f;
typedef __attribute__((ext_vector_type(8)))  __bf16 v8bf;
typedef __attribute__((ext_vector_type(16))) __bf16 v16bf;

static constexpr int Tn = 8192;
static constexpr int Hn = 2048;
static constexpr int In = 4096;
static constexpr int En = 8;

static constexpr int BM  = 128;
static constexpr int BN  = 64;
static constexpr int BK  = 32;
static constexpr int LDK = 40;   // padded LDS k-stride (halfs); 80B rows, 16B-aligned

// Workspace layout (bytes); requires ws_size >= ~161 MiB
static constexpr size_t OFF_COUNTS  = 0;                       // E ints
static constexpr size_t OFF_BASES   = 256;                     // E ints
static constexpr size_t OFF_TOPKID  = 512;                     // 2T ints
static constexpr size_t OFF_TOPKW   = OFF_TOPKID + 2ull*Tn*4;  // 2T floats
static constexpr size_t OFF_TOKLIST = OFF_TOPKW  + 2ull*Tn*4;  // 2T ints
static constexpr size_t OFF_TOKW    = OFF_TOKLIST+ 2ull*Tn*4;  // 2T floats
static constexpr size_t OFF_XBF     = OFF_TOKW   + 2ull*Tn*4;  // T*H bf16
static constexpr size_t OFF_HBF     = OFF_XBF    + 2ull*Tn*Hn; // 2T*I bf16

// ---- CDNA5 async global->LDS copy (16B), tracked by ASYNCcnt --------------
__device__ __forceinline__ void async_ld16(void* lds_ptr, const void* gptr) {
  unsigned loff = (unsigned)(size_t)lds_ptr;                 // flat[31:0] = LDS offset
  unsigned long long ga = (unsigned long long)(size_t)gptr; // global address
  asm volatile("global_load_async_to_lds_b128 %0, %1, off"
               :: "v"(loff), "v"(ga) : "memory");
}
__device__ __forceinline__ void wait_async0() {
  asm volatile("s_wait_asynccnt 0" ::: "memory");
}

__device__ __forceinline__ v16bf ld_frag16(const __bf16* p0, const __bf16* p1) {
  v8bf a = *(const v8bf*)p0;   // ds_load_b128
  v8bf b = *(const v8bf*)p1;   // ds_load_b128
  v16bf r;
#pragma unroll
  for (int i = 0; i < 8; ++i) { r[i] = a[i]; r[i + 8] = b[i]; }
  return r;
}

__device__ __forceinline__ void cvt_store8(__bf16* dst, const float* src) {
  float4 f0 = ((const float4*)src)[0];
  float4 f1 = ((const float4*)src)[1];
  v8bf o;
  o[0] = (__bf16)f0.x; o[1] = (__bf16)f0.y; o[2] = (__bf16)f0.z; o[3] = (__bf16)f0.w;
  o[4] = (__bf16)f1.x; o[5] = (__bf16)f1.y; o[6] = (__bf16)f1.z; o[7] = (__bf16)f1.w;
  *(v8bf*)dst = o;             // ds_store_b128
}

// ---------------- 1) zero output + counters --------------------------------
__global__ __launch_bounds__(256) void k_zero(float* out, int* counts) {
  size_t gid = (size_t)blockIdx.x * blockDim.x + threadIdx.x;
  size_t idx = gid * 4;
  if (idx + 3 < (size_t)Tn * Hn) {
    float4 z = {0.f, 0.f, 0.f, 0.f};
    *(float4*)(out + idx) = z;
  }
  if (gid < En) counts[gid] = 0;
}

// ---------------- 2) x fp32 -> bf16 ----------------------------------------
__global__ __launch_bounds__(256) void k_cvt_x(const float* __restrict__ x,
                                               __bf16* __restrict__ xb) {
  size_t gid = (size_t)blockIdx.x * blockDim.x + threadIdx.x;
  size_t idx = gid * 8;
  if (idx + 7 < (size_t)Tn * Hn) cvt_store8(xb + idx, x + idx);
}

// ---------------- 3) gating: logits, softmax, top-2 ------------------------
__global__ __launch_bounds__(256) void k_gate(const float* __restrict__ x,
                                              const float* __restrict__ gw,
                                              int* __restrict__ topk_ids,
                                              float* __restrict__ topk_w,
                                              int* __restrict__ counts) {
  int lane = threadIdx.x & 31, wave = threadIdx.x >> 5;
  int t = blockIdx.x * 8 + wave;
  if (t >= Tn) return;
  const float* xr = x + (size_t)t * Hn;
  float xv[Hn / 32];
#pragma unroll
  for (int k = 0; k < Hn / 32; ++k) xv[k] = xr[lane + 32 * k];
  float logit[En];
  for (int e = 0; e < En; ++e) {
    const float* g = gw + (size_t)e * Hn;
    float s = 0.f;
#pragma unroll
    for (int k = 0; k < Hn / 32; ++k) s += xv[k] * g[lane + 32 * k];
#pragma unroll
    for (int off = 16; off > 0; off >>= 1) s += __shfl_xor(s, off, 32);
    logit[e] = s;
  }
  if (lane == 0) {
    float m = logit[0];
    for (int e = 1; e < En; ++e) m = fmaxf(m, logit[e]);
    float p[En];
    for (int e = 0; e < En; ++e) p[e] = __expf(logit[e] - m);
    int i0 = 0;
    for (int e = 1; e < En; ++e) if (p[e] > p[i0]) i0 = e;
    int i1 = (i0 == 0) ? 1 : 0;
    for (int e = 0; e < En; ++e) if (e != i0 && p[e] > p[i1]) i1 = e;
    float sw = p[i0] + p[i1];
    topk_ids[2 * t] = i0; topk_ids[2 * t + 1] = i1;
    topk_w[2 * t] = p[i0] / sw; topk_w[2 * t + 1] = p[i1] / sw;
    atomicAdd(&counts[i0], 1);
    atomicAdd(&counts[i1], 1);
  }
}

// ---------------- 4) exclusive prefix of counts ----------------------------
__global__ void k_prefix(const int* __restrict__ counts, int* __restrict__ bases) {
  if (threadIdx.x == 0 && blockIdx.x == 0) {
    int run = 0;
    for (int e = 0; e < En; ++e) { bases[e] = run; run += counts[e]; }
  }
}

// ---------------- 5) deterministic per-expert compaction -------------------
__global__ __launch_bounds__(256) void k_compact(const int* __restrict__ topk_ids,
                                                 const float* __restrict__ topk_w,
                                                 const int* __restrict__ bases,
                                                 int* __restrict__ tok_list,
                                                 float* __restrict__ tok_w) {
  __shared__ int flags[256];
  __shared__ int pref[257];
  __shared__ int basev;
  const int e = blockIdx.x;
  const int tid = threadIdx.x;
  if (tid == 0) basev = bases[e];
  __syncthreads();
  for (int chunk = 0; chunk < Tn / 256; ++chunk) {
    int t = chunk * 256 + tid;
    int flag = 0; float w = 0.f;
    int a = topk_ids[2 * t], b = topk_ids[2 * t + 1];
    if (a == e)      { flag = 1; w = topk_w[2 * t]; }
    else if (b == e) { flag = 1; w = topk_w[2 * t + 1]; }
    flags[tid] = flag;
    __syncthreads();
    if (tid == 0) {
      int run = 0;
      for (int i = 0; i < 256; ++i) { pref[i] = run; run += flags[i]; }
      pref[256] = run;
    }
    __syncthreads();
    if (flag) {
      int pos = basev + pref[tid];
      tok_list[pos] = t; tok_w[pos] = w;
    }
    __syncthreads();
    if (tid == 0) basev += pref[256];
    __syncthreads();
  }
}

// ---------------- 6) GEMM1 fused: h = silu(x wsg^T) * (x wsu^T) ------------
__global__ __launch_bounds__(256) void k_gemm1(const __bf16* __restrict__ xb,
                                               const float* __restrict__ wts,
                                               const int* __restrict__ counts,
                                               const int* __restrict__ bases,
                                               const int* __restrict__ tok_list,
                                               __bf16* __restrict__ hbuf) {
  const int e = blockIdx.z;
  const int cnt = counts[e];
  const int m0 = blockIdx.x * BM;
  if (m0 >= cnt) return;
  const int n0 = blockIdx.y * BN;
  const int base = bases[e];
  const float* wsE = wts + (size_t)e * (2ull * In) * Hn;

  __shared__ __bf16 As[2][BM][LDK];
  __shared__ __bf16 Bg[2][BN][LDK];
  __shared__ __bf16 Bu[2][BN][LDK];
  __shared__ int toks[BM];

  const int tid = threadIdx.x;
  if (tid < BM) {
    int slot = m0 + tid;
    toks[tid] = tok_list[base + (slot < cnt ? slot : 0)];
  }
  __syncthreads();

  const int lane = tid & 31, wave = tid >> 5;
  const int wm = (wave >> 1) * 32, wn = (wave & 1) * 32;

  v8f cg[2][2], cu[2][2];
#pragma unroll
  for (int mi = 0; mi < 2; ++mi)
#pragma unroll
    for (int ni = 0; ni < 2; ++ni)
#pragma unroll
      for (int r = 0; r < 8; ++r) { cg[mi][ni][r] = 0.f; cu[mi][ni][r] = 0.f; }

  const int arow = lane & 15, kb  = (lane >> 4) * 8;
  const int bcol = lane & 15, kb2 = (lane >> 4) * 8;

  // A rows: 128 rows x 32 halfs = 512 x 16B chunks; 2 async copies per thread
  const int ar = (tid + 0)   >> 2, asg = (tid + 0)   & 3;
  const int br = (tid + 256) >> 2, bsg = (tid + 256) & 3;
  const int wr = tid >> 2,         wsg = tid & 3;

  auto stage = [&](int buf, int k0) {
    async_ld16(&As[buf][ar][asg * 8],
               xb + (size_t)toks[ar] * Hn + k0 + asg * 8);
    async_ld16(&As[buf][br][bsg * 8],
               xb + (size_t)toks[br] * Hn + k0 + bsg * 8);
    cvt_store8(&Bg[buf][wr][wsg * 8],
               wsE + (size_t)(n0 + wr) * Hn + k0 + wsg * 8);
    cvt_store8(&Bu[buf][wr][wsg * 8],
               wsE + (size_t)(In + n0 + wr) * Hn + k0 + wsg * 8);
  };

  auto compute = [&](int buf) {
    v16bf a[2], bg[2], bu[2];
#pragma unroll
    for (int mi = 0; mi < 2; ++mi)
      a[mi] = ld_frag16(&As[buf][wm + mi * 16 + arow][kb],
                        &As[buf][wm + mi * 16 + arow][kb + 16]);
#pragma unroll
    for (int ni = 0; ni < 2; ++ni) {
      bg[ni] = ld_frag16(&Bg[buf][wn + ni * 16 + bcol][kb2],
                         &Bg[buf][wn + ni * 16 + bcol][kb2 + 16]);
      bu[ni] = ld_frag16(&Bu[buf][wn + ni * 16 + bcol][kb2],
                         &Bu[buf][wn + ni * 16 + bcol][kb2 + 16]);
    }
#pragma unroll
    for (int mi = 0; mi < 2; ++mi)
#pragma unroll
      for (int ni = 0; ni < 2; ++ni) {
        cg[mi][ni] = __builtin_amdgcn_wmma_f32_16x16x32_bf16(
            false, a[mi], false, bg[ni], (short)0, cg[mi][ni], false, false);
        cu[mi][ni] = __builtin_amdgcn_wmma_f32_16x16x32_bf16(
            false, a[mi], false, bu[ni], (short)0, cu[mi][ni], false, false);
      }
  };

  stage(0, 0);
  wait_async0();
  __syncthreads();
  constexpr int NK = Hn / BK;
  for (int kk = 0; kk < NK; ++kk) {
    int p = kk & 1;
    if (kk + 1 < NK) stage(1 - p, (kk + 1) * BK);  // overlap with WMMAs below
    compute(p);
    wait_async0();       // this wave's async LDS writes landed
    __syncthreads();     // (compiler adds s_wait_dscnt before the signal)
  }

  // epilogue: h = silu(g)*u, scatter to hbuf (bf16)
  const int rlo = (lane >> 4) * 8, ncol = lane & 15;
#pragma unroll
  for (int mi = 0; mi < 2; ++mi)
#pragma unroll
    for (int ni = 0; ni < 2; ++ni)
#pragma unroll
      for (int r = 0; r < 8; ++r) {
        int m = wm + mi * 16 + rlo + r;
        int slot = m0 + m;
        if (slot < cnt) {
          float g = cg[mi][ni][r], u = cu[mi][ni][r];
          float h = (g / (1.f + __expf(-g))) * u;
          hbuf[(size_t)(base + slot) * In + (n0 + wn + ni * 16 + ncol)] =
              (__bf16)h;
        }
      }
}

// ---------------- 7) GEMM2: out += cw * (h w2^T) ---------------------------
__global__ __launch_bounds__(256) void k_gemm2(const __bf16* __restrict__ hbuf,
                                               const float* __restrict__ w2,
                                               const int* __restrict__ counts,
                                               const int* __restrict__ bases,
                                               const int* __restrict__ tok_list,
                                               const float* __restrict__ tok_w,
                                               float* __restrict__ out) {
  const int e = blockIdx.z;
  const int cnt = counts[e];
  const int m0 = blockIdx.x * BM;
  if (m0 >= cnt) return;
  const int n0 = blockIdx.y * BN;
  const int base = bases[e];
  const float* w2E = w2 + (size_t)e * Hn * In;

  __shared__ __bf16 As[2][BM][LDK];
  __shared__ __bf16 Bs[2][BN][LDK];
  __shared__ int   toks[BM];
  __shared__ float wts[BM];

  const int tid = threadIdx.x;
  if (tid < BM) {
    int slot = m0 + tid;
    int cs = slot < cnt ? slot : cnt - 1;
    toks[tid] = tok_list[base + cs];
    wts[tid]  = tok_w[base + cs];
  }
  __syncthreads();

  const int lane = tid & 31, wave = tid >> 5;
  const int wm = (wave >> 1) * 32, wn = (wave & 1) * 32;

  v8f cc[2][2];
#pragma unroll
  for (int mi = 0; mi < 2; ++mi)
#pragma unroll
    for (int ni = 0; ni < 2; ++ni)
#pragma unroll
      for (int r = 0; r < 8; ++r) cc[mi][ni][r] = 0.f;

  const int arow = lane & 15, kb  = (lane >> 4) * 8;
  const int bcol = lane & 15, kb2 = (lane >> 4) * 8;

  const int ar = (tid + 0)   >> 2, asg = (tid + 0)   & 3;
  const int br = (tid + 256) >> 2, bsg = (tid + 256) & 3;
  const int wr = tid >> 2,         wsg = tid & 3;
  const int arS = m0 + ar < cnt ? m0 + ar : cnt - 1;  // clamp to stay in hbuf
  const int brS = m0 + br < cnt ? m0 + br : cnt - 1;

  auto stage = [&](int buf, int k0) {
    async_ld16(&As[buf][ar][asg * 8],
               hbuf + (size_t)(base + arS) * In + k0 + asg * 8);
    async_ld16(&As[buf][br][bsg * 8],
               hbuf + (size_t)(base + brS) * In + k0 + bsg * 8);
    cvt_store8(&Bs[buf][wr][wsg * 8],
               w2E + (size_t)(n0 + wr) * In + k0 + wsg * 8);
  };

  auto compute = [&](int buf) {
    v16bf a[2], b[2];
#pragma unroll
    for (int mi = 0; mi < 2; ++mi)
      a[mi] = ld_frag16(&As[buf][wm + mi * 16 + arow][kb],
                        &As[buf][wm + mi * 16 + arow][kb + 16]);
#pragma unroll
    for (int ni = 0; ni < 2; ++ni)
      b[ni] = ld_frag16(&Bs[buf][wn + ni * 16 + bcol][kb2],
                        &Bs[buf][wn + ni * 16 + bcol][kb2 + 16]);
#pragma unroll
    for (int mi = 0; mi < 2; ++mi)
#pragma unroll
      for (int ni = 0; ni < 2; ++ni)
        cc[mi][ni] = __builtin_amdgcn_wmma_f32_16x16x32_bf16(
            false, a[mi], false, b[ni], (short)0, cc[mi][ni], false, false);
  };

  stage(0, 0);
  wait_async0();
  __syncthreads();
  constexpr int NK = In / BK;
  for (int kk = 0; kk < NK; ++kk) {
    int p = kk & 1;
    if (kk + 1 < NK) stage(1 - p, (kk + 1) * BK);
    compute(p);
    wait_async0();
    __syncthreads();
  }

  const int rlo = (lane >> 4) * 8, ncol = lane & 15;
#pragma unroll
  for (int mi = 0; mi < 2; ++mi)
#pragma unroll
    for (int ni = 0; ni < 2; ++ni)
#pragma unroll
      for (int r = 0; r < 8; ++r) {
        int m = wm + mi * 16 + rlo + r;
        int slot = m0 + m;
        if (slot < cnt) {
          float v = cc[mi][ni][r] * wts[m];
          atomicAdd(&out[(size_t)toks[m] * Hn + (n0 + wn + ni * 16 + ncol)], v);
        }
      }
}

// ---------------------------------------------------------------------------
extern "C" void kernel_launch(void* const* d_in, const int* in_sizes, int n_in,
                              void* d_out, int out_size, void* d_ws, size_t ws_size,
                              hipStream_t stream) {
  const float* x    = (const float*)d_in[0];
  const float* gw   = (const float*)d_in[1];
  const float* wts  = (const float*)d_in[2];   // ws (E, 2I, H)
  const float* w2   = (const float*)d_in[3];   // w2s (E, H, I)
  float* out = (float*)d_out;

  char* w = (char*)d_ws;
  int*    counts   = (int*)   (w + OFF_COUNTS);
  int*    bases    = (int*)   (w + OFF_BASES);
  int*    topk_ids = (int*)   (w + OFF_TOPKID);
  float*  topk_w   = (float*) (w + OFF_TOPKW);
  int*    tok_list = (int*)   (w + OFF_TOKLIST);
  float*  tok_w    = (float*) (w + OFF_TOKW);
  __bf16* xb       = (__bf16*)(w + OFF_XBF);
  __bf16* hbuf     = (__bf16*)(w + OFF_HBF);

  const size_t TH = (size_t)Tn * Hn;
  k_zero <<<(TH / 4 + 255) / 256, 256, 0, stream>>>(out, counts);
  k_cvt_x<<<(TH / 8 + 255) / 256, 256, 0, stream>>>(x, xb);
  k_gate <<<Tn / 8, 256, 0, stream>>>(x, gw, topk_ids, topk_w, counts);
  k_prefix<<<1, 1, 0, stream>>>(counts, bases);
  k_compact<<<En, 256, 0, stream>>>(topk_ids, topk_w, bases, tok_list, tok_w);
  k_gemm1<<<dim3(Tn / BM, In / BN, En), 256, 0, stream>>>(
      xb, wts, counts, bases, tok_list, hbuf);
  k_gemm2<<<dim3(Tn / BM, Hn / BN, En), 256, 0, stream>>>(
      hbuf, w2, counts, bases, tok_list, tok_w, out);
}
